// GraphSAGE_50723563765895
// MI455X (gfx1250) — matrix operand order repaired and checked by reading, not compiled, not run on hardware
//
#include <hip/hip_runtime.h>
#include <stdint.h>

typedef __attribute__((ext_vector_type(16))) _Float16 v16h;
typedef __attribute__((ext_vector_type(8)))  _Float16 v8h;
typedef __attribute__((ext_vector_type(8)))  float    v8f;

#define N_NODES 50000
#define HID     128
#define KTOT    256   // concatenated K: [agg | h_self]
#define NT      8     // 128 / 16 output col tiles
#define KS      8     // 256 / 32 K steps
#define PACK_HALVES_PER_LAYER (KS * NT * 32 * 16)  // 32768

__global__ void zero_kernel(float* p, int n) {
  int i = blockIdx.x * blockDim.x + threadIdx.x;
  if (i < n) p[i] = 0.0f;
}

__global__ void degree_kernel(const long long* __restrict__ dst, float* deg, int nE) {
  int e = blockIdx.x * blockDim.x + threadIdx.x;
  if (e < nE) atomicAdd(&deg[(int)dst[e]], 1.0f);
}

__global__ void invdeg_kernel(float* deg, int n) {
  int i = blockIdx.x * blockDim.x + threadIdx.x;
  if (i < n) deg[i] = 1.0f / fmaxf(deg[i], 1.0f);
}

// One 32-lane group per edge; 4 features per lane. Coalesced 512B gather from
// h[src], f32 global atomic adds into agg[dst]. This phase is the HBM roofline.
__global__ void scatter_kernel(const float* __restrict__ h,
                               const long long* __restrict__ src,
                               const long long* __restrict__ dst,
                               float* __restrict__ agg, int nE) {
  int g = blockIdx.x * blockDim.x + threadIdx.x;
  int e = g >> 5;
  int l = g & 31;
  if (e >= nE) return;
  const float* hs = h + (size_t)((int)src[e]) * HID;
  float*       ad = agg + (size_t)((int)dst[e]) * HID;
#pragma unroll
  for (int c = 0; c < 4; ++c) {
    int f = l + c * 32;
    atomicAdd(&ad[f], hs[f]);
  }
}

// Pre-swizzle [Wl;Wr] (256x128 f32, row-major fan_in x fan_out) into the WMMA
// B-fragment layout (contiguous K order, not the ISA table; 16-bit B 32x16): for fragment (s,t), lane l,
// element e:  N = t*16 + (l&15),  K = s*32 + (l>>4)*16 + e.
// Stored lane-major so the GEMM reads one contiguous 32B vector per lane.
__global__ void pack_weights_kernel(const float* __restrict__ W1l, const float* __restrict__ W1r,
                                    const float* __restrict__ W2l, const float* __restrict__ W2r,
                                    _Float16* __restrict__ packed) {
  int id = blockIdx.x * blockDim.x + threadIdx.x;  // 65536 total
  int e = id & 15;
  int l = (id >> 4) & 31;
  int t = (id >> 9) & 7;
  int s = (id >> 12) & 7;
  int layer = id >> 15;
  int n = t * 16 + (l & 15);
  int k = s * 32 + (l >> 4) * 16 + e;
  const float* Wl = layer ? W2l : W1l;
  const float* Wr = layer ? W2r : W1r;
  float v = (k < HID) ? Wl[k * HID + n] : Wr[(k - HID) * HID + n];
  packed[(size_t)layer * PACK_HALVES_PER_LAYER + (((s * NT + t) * 32 + l) * 16 + e)]
      = (_Float16)v;
}

// out[r0..r0+15][0..127] = relu( [agg*invdeg | xself] @ packW + bias )
// 256 threads = 8 waves; wave w owns N-tile w; 8 x v_wmma_f32_16x16x32_f16.
__global__ void __launch_bounds__(256)
sage_gemm_kernel(const float* __restrict__ agg,
                 const float* __restrict__ invdeg,
                 const float* __restrict__ xself,
                 const _Float16* __restrict__ packW,
                 const float* __restrict__ bias,
                 float* __restrict__ out) {
  __shared__ __align__(16) _Float16 lA[16 * KTOT];  // 8 KB A panel, f16

  const int r0  = blockIdx.x * 16;
  const int tid = threadIdx.x;

  // Stage A panel: rows r0..r0+15, K = [0,128): mean-agg, [128,256): self.
#pragma unroll
  for (int i = 0; i < 16; ++i) {
    int idx = tid + i * 256;       // 0..4095
    int row = idx >> 8;
    int col = idx & 255;
    int r   = r0 + row;
    float v;
    if (col < HID) v = agg[(size_t)r * HID + col] * invdeg[r];
    else           v = xself[(size_t)r * HID + (col - HID)];
    lA[idx] = (_Float16)v;
  }
  __syncthreads();

  const int w    = tid >> 5;   // wave id == N tile
  const int l    = tid & 31;
  const int m    = l & 15;     // A row / C col within tile
  const int half = l >> 4;

  v8f c = {};
#pragma unroll
  for (int s = 0; s < KS; ++s) {
    // A fragment (contiguous K order, matching packed B): e -> K=s*32+half*16+e.
    // Both halves are contiguous 16B in LDS.
    const _Float16* arow = &lA[m * KTOT + s * 32 + half * 16];
    v8h lo = *(const v8h*)(arow);
    v8h hi = *(const v8h*)(arow + 8);
    v16h a;
#pragma unroll
    for (int e2 = 0; e2 < 8; ++e2) { a[e2] = lo[e2]; a[e2 + 8] = hi[e2]; }

    // B fragment: contiguous 32B per lane from the pre-swizzled buffer.
    v16h b = *(const v16h*)(packW + (((size_t)(s * NT + w) * 32 + l) << 4));

    c = __builtin_amdgcn_wmma_f32_16x16x32_f16(false, a, false, b,
                                               (short)0, c, false, false);
  }

  // C/D layout: element j -> row r0 + half*8 + j, col w*16 + m.
  const int   n  = w * 16 + m;
  const float bn = bias[n];
#pragma unroll
  for (int j = 0; j < 8; ++j) {
    int row = r0 + half * 8 + j;
    out[(size_t)row * HID + n] = fmaxf(c[j] + bn, 0.0f);
  }
}

__global__ void fc_kernel(const float* __restrict__ h, const float* __restrict__ Wfc,
                          const float* __restrict__ bfc, float* __restrict__ out, int n) {
  int i = blockIdx.x * blockDim.x + threadIdx.x;
  if (i >= n) return;
  float a0 = bfc[0], a1 = bfc[1];
  const float* hr = h + (size_t)i * HID;
#pragma unroll 4
  for (int k = 0; k < HID; ++k) {
    float v = hr[k];
    a0 += v * Wfc[k * 2 + 0];
    a1 += v * Wfc[k * 2 + 1];
  }
  out[i * 2 + 0] = a0;
  out[i * 2 + 1] = a1;
}

extern "C" void kernel_launch(void* const* d_in, const int* in_sizes, int n_in,
                              void* d_out, int out_size, void* d_ws, size_t ws_size,
                              hipStream_t stream) {
  const float*     x    = (const float*)d_in[0];
  const long long* eidx = (const long long*)d_in[1];   // int64, shape (2, nE)
  const float*     W1l  = (const float*)d_in[2];
  const float*     W1r  = (const float*)d_in[3];
  const float*     b1   = (const float*)d_in[4];
  const float*     W2l  = (const float*)d_in[5];
  const float*     W2r  = (const float*)d_in[6];
  const float*     b2   = (const float*)d_in[7];
  const float*     Wfc  = (const float*)d_in[8];
  const float*     bfc  = (const float*)d_in[9];

  const int nE = in_sizes[1] / 2;
  const long long* srcI = eidx;        // row 0: message sources
  const long long* dstI = eidx + nE;   // row 1: aggregation targets

  // Workspace layout (floats): [deg 50176][agg 6.4M][h 6.4M][packW f16 64K halves]
  float* ws  = (float*)d_ws;
  const size_t DEGF = 50176;
  float* deg = ws;
  float* agg = ws + DEGF;
  float* h   = agg + (size_t)N_NODES * HID;
  _Float16* packW = (_Float16*)(h + (size_t)N_NODES * HID);

  const int TPB = 256;
  const int nScatterBlk = (int)(((size_t)nE * 32 + TPB - 1) / TPB);

  // degrees + inverse degrees
  { int n = (int)(DEGF + (size_t)N_NODES * HID);
    zero_kernel<<<(n + TPB - 1) / TPB, TPB, 0, stream>>>(ws, n); }
  degree_kernel<<<(nE + TPB - 1) / TPB, TPB, 0, stream>>>(dstI, deg, nE);
  invdeg_kernel<<<(N_NODES + TPB - 1) / TPB, TPB, 0, stream>>>(deg, N_NODES);

  // pack weights for both SAGE layers into WMMA B-fragment order
  pack_weights_kernel<<<(2 * PACK_HALVES_PER_LAYER) / TPB, TPB, 0, stream>>>(
      W1l, W1r, W2l, W2r, packW);

  // layer 1
  scatter_kernel<<<nScatterBlk, TPB, 0, stream>>>(x, srcI, dstI, agg, nE);
  sage_gemm_kernel<<<N_NODES / 16, TPB, 0, stream>>>(agg, deg, x, packW, b1, h);

  // layer 2 (in-place over h; each workgroup stages its rows to LDS first)
  { int n = (int)((size_t)N_NODES * HID);
    zero_kernel<<<(n + TPB - 1) / TPB, TPB, 0, stream>>>(agg, n); }
  scatter_kernel<<<nScatterBlk, TPB, 0, stream>>>(h, srcI, dstI, agg, nE);
  sage_gemm_kernel<<<N_NODES / 16, TPB, 0, stream>>>(agg, deg, h,
                                                     packW + PACK_HALVES_PER_LAYER, b2, h);

  // classifier head
  fc_kernel<<<(N_NODES + TPB - 1) / TPB, TPB, 0, stream>>>(h, Wfc, bfc,
                                                           (float*)d_out, N_NODES);
}
